// GroupRouter_9234179687133
// MI455X (gfx1250) — matrix operand order repaired and checked by reading, not compiled, not run on hardware
//
#include <hip/hip_runtime.h>
#include <hip/hip_bf16.h>
#include <math.h>

// ---------------------------------------------------------------------------
// CDNA5 WMMA types
// ---------------------------------------------------------------------------
typedef __attribute__((ext_vector_type(16))) __bf16 v16bf;
typedef __attribute__((ext_vector_type(8)))  float  v8f;

union Frag {
  v16bf v;
  uint4 q[2];
  unsigned short h[16];
};

__device__ __forceinline__ unsigned short f2bf(float f) {
  unsigned int u = __float_as_uint(f);
  u += 0x7fffu + ((u >> 16) & 1u);   // round-to-nearest-even
  return (unsigned short)(u >> 16);
}

__device__ __forceinline__ float gelu_exact(float x) {
  return 0.5f * x * (1.0f + erff(x * 0.7071067811865475f));
}
__device__ __forceinline__ float sigmoid_(float x) {
  return 1.0f / (1.0f + expf(-x));
}

__device__ __forceinline__ void loadfrag(Frag& f, const unsigned short* p) {
  const uint4* q = (const uint4*)p;
  f.q[0] = q[0];
  f.q[1] = q[1];
}

// Inverse of the 16-bit A-matrix (16x32) fragment swizzle (ISA 7.12.2):
// forward: lane L (half = L>>4, m = L&15), element e -> K = e + 8*half + (e>=8 ? 8 : 0)
__device__ __forceinline__ void inv_amap(int ko, int m, int& lane, int& e) {
  int q    = ko >> 3;       // 0..3
  int half = q & 1;
  e    = ko - 8 * half - 8 * (q >> 1);
  lane = m + 16 * half;
}

// ---------------------------------------------------------------------------
// Convert x [B,D] f32 row-major -> bf16 A-fragment layout
// layout: [rowgroup rg][kstep ks][lane 0..31][halfword 0..15]
// ---------------------------------------------------------------------------
__global__ __launch_bounds__(256) void cvt_x_kernel(const float* __restrict__ x,
                                                    unsigned short* __restrict__ xf,
                                                    int KS, long total) {
  long t = (long)blockIdx.x * 256 + threadIdx.x;
  if (t >= total) return;
  int  e    = (int)(t & 15);
  int  lane = (int)((t >> 4) & 31);
  long r    = t >> 9;
  int  ks   = (int)(r % KS);
  long rg   = r / KS;
  long b    = rg * 16 + (lane & 15);
  int  k    = ks * 32 + e + 8 * (lane >> 4) + ((e >= 8) ? 8 : 0);
  xf[t] = f2bf(x[b * (long)(KS * 32) + k]);
}

// ---------------------------------------------------------------------------
// Convert weight [K,N] f32 row-major -> bf16 B-fragment layout
// layout: [kstep ks][nfrag nf][lane][halfword]; K = e + 8*(lane>>4) + (e>=8 ? 8 : 0), N = (lane&15)
// ---------------------------------------------------------------------------
__global__ __launch_bounds__(256) void cvt_w_kernel(const float* __restrict__ w,
                                                    unsigned short* __restrict__ wf,
                                                    int N, long total) {
  long t = (long)blockIdx.x * 256 + threadIdx.x;
  if (t >= total) return;
  int  e    = (int)(t & 15);
  int  lane = (int)((t >> 4) & 31);
  long r    = t >> 9;
  int  NF   = N >> 4;
  int  nf   = (int)(r % NF);
  int  ks   = (int)(r / NF);
  int  k    = ks * 32 + e + 8 * (lane >> 4) + ((e >= 8) ? 8 : 0);
  int  n    = nf * 16 + (lane & 15);
  wf[t] = f2bf(w[(long)k * N + n]);
}

// ---------------------------------------------------------------------------
// bf16 WMMA GEMM, software-pipelined (double-buffered fragments):
//   out = ACT(Afrag @ Wfrag + bias)
//   block = 256 threads = 8 waves; each wave: one 16-row group x 4 n-fragments
//   grid.x = M/128, grid.y = N/64 ; KS must be even
// ACT: 0=none 1=gelu 2=sigmoid ; OUT: 0=f32 row-major, 1=bf16 A-fragment layout
// ---------------------------------------------------------------------------
template <int ACT, int OUT>
__global__ __launch_bounds__(256) void gemm_bf16(const unsigned short* __restrict__ Af,
                                                 const unsigned short* __restrict__ Wf,
                                                 const float* __restrict__ bias,
                                                 void* __restrict__ outp,
                                                 int KS, int N) {
  const int  lane = threadIdx.x & 31;
  const int  wave = threadIdx.x >> 5;
  const int  NF   = N >> 4;
  const long rg   = (long)blockIdx.x * 8 + wave;
  const int  nf0  = blockIdx.y * 4;

  v8f acc[4];
#pragma unroll
  for (int j = 0; j < 4; ++j)
#pragma unroll
    for (int i = 0; i < 8; ++i) acc[j][i] = 0.0f;

  const long wstep = (long)NF * 512;               // shorts per kstep in Wf
  const unsigned short* aptr = Af + ((rg * KS) * 32 + lane) * 16;
  const unsigned short* wptr = Wf + ((long)nf0 * 32 + lane) * 16;

  Frag A0, A1, W0[4], W1[4];

  // prologue: stage kstep 0 into buffer 0
  loadfrag(A0, aptr);
#pragma unroll
  for (int j = 0; j < 4; ++j) loadfrag(W0[j], wptr + (long)j * 512);

  for (int ks = 0; ks < KS; ks += 2) {
    // preload kstep ks+1 into buffer 1 (KS even -> always exists)
    loadfrag(A1, aptr + 512);
#pragma unroll
    for (int j = 0; j < 4; ++j) loadfrag(W1[j], wptr + wstep + (long)j * 512);
    if (ks + 2 < KS) __builtin_prefetch(aptr + 1024, 0, 1);

#pragma unroll
    for (int j = 0; j < 4; ++j)
      acc[j] = __builtin_amdgcn_wmma_f32_16x16x32_bf16(
          false, A0.v, false, W0[j].v, (short)0, acc[j], false, false);

    aptr += 1024;
    wptr += 2 * wstep;

    // preload kstep ks+2 into buffer 0 while buffer-1 WMMAs run
    if (ks + 2 < KS) {
      loadfrag(A0, aptr);
#pragma unroll
      for (int j = 0; j < 4; ++j) loadfrag(W0[j], wptr + (long)j * 512);
    }

#pragma unroll
    for (int j = 0; j < 4; ++j)
      acc[j] = __builtin_amdgcn_wmma_f32_16x16x32_bf16(
          false, A1.v, false, W1[j].v, (short)0, acc[j], false, false);
  }

  const int mlhi = 8 * (lane >> 4);
  const int nl   = lane & 15;
#pragma unroll
  for (int j = 0; j < 4; ++j) {
    int   col = (nf0 + j) * 16 + nl;
    float bv  = bias[col];
#pragma unroll
    for (int r = 0; r < 8; ++r) {
      int   m   = r + mlhi;                 // 0..15 inside row group
      long  row = rg * 16 + m;
      float v   = acc[j][r] + bv;
      if (ACT == 1) v = gelu_exact(v);
      else if (ACT == 2) v = sigmoid_(v);
      if (OUT == 0) {
        ((float*)outp)[row * N + col] = v;
      } else {
        int ks2 = col >> 5, ko = col & 31, l2, e2;
        inv_amap(ko, m, l2, e2);
        ((unsigned short*)outp)[((rg * (long)(N >> 5) + ks2) * 32 + l2) * 16 + e2] = f2bf(v);
      }
    }
  }
}

// ---------------------------------------------------------------------------
// LayerNorm(256) + GELU, one wave32 per row; writes bf16 A-fragment layout
// ---------------------------------------------------------------------------
__global__ __launch_bounds__(256) void ln_gelu_kernel(const float* __restrict__ y,
                                                      const float* __restrict__ gam,
                                                      const float* __restrict__ bet,
                                                      unsigned short* __restrict__ h1f) {
  int  wave = threadIdx.x >> 5, lane = threadIdx.x & 31;
  long b = (long)blockIdx.x * 8 + wave;
  const float* row = y + b * 256;
  float v[8], s = 0.0f, s2 = 0.0f;
#pragma unroll
  for (int i = 0; i < 8; ++i) {
    v[i] = row[lane * 8 + i];
    s += v[i];
    s2 += v[i] * v[i];
  }
#pragma unroll
  for (int m = 16; m > 0; m >>= 1) {
    s  += __shfl_xor(s, m, 32);
    s2 += __shfl_xor(s2, m, 32);
  }
  float mean = s * (1.0f / 256.0f);
  float var  = s2 * (1.0f / 256.0f) - mean * mean;
  float inv  = rsqrtf(var + 1e-5f);
  long rg = b >> 4;
  int  m8 = (int)(b & 15);
#pragma unroll
  for (int i = 0; i < 8; ++i) {
    int   col = lane * 8 + i;
    float h   = (v[i] - mean) * inv * gam[col] + bet[col];
    h = gelu_exact(h);
    int ks = col >> 5, ko = col & 31, l2, e2;
    inv_amap(ko, m8, l2, e2);
    h1f[((rg * 8 + ks) * 32 + l2) * 16 + e2] = f2bf(h);
  }
}

// ---------------------------------------------------------------------------
// logits = h2 @ w3 + b3 + gi ; probs = sigmoid ; constrained hard selection;
// deterministic per-block tree reductions -> pblock[block*10 + v]
//   v: 0=sum(#probs>0.5), 1..8=sum(probs_g>0.5), 9=sum(probs)
// ---------------------------------------------------------------------------
__global__ __launch_bounds__(256) void router_final_kernel(const float* __restrict__ h2,
                                                           const float* __restrict__ w3,
                                                           const float* __restrict__ b3,
                                                           const float* __restrict__ gi,
                                                           float* __restrict__ out_probs,
                                                           float* __restrict__ out_sel,
                                                           float* __restrict__ pblock) {
  __shared__ float s_w3[128 * 8];
  __shared__ float s_b[8], s_g[8];
  __shared__ float sr[10 * 256];
  int tid = threadIdx.x;
  for (int i = tid; i < 1024; i += 256) s_w3[i] = w3[i];
  if (tid < 8) { s_b[tid] = b3[tid]; s_g[tid] = gi[tid]; }
  __syncthreads();

  long b = (long)blockIdx.x * 256 + tid;
  float logit[8];
#pragma unroll
  for (int g = 0; g < 8; ++g) logit[g] = 0.0f;
  const float* hrow = h2 + b * 128;
  for (int k = 0; k < 128; ++k) {
    float v = hrow[k];
#pragma unroll
    for (int g = 0; g < 8; ++g) logit[g] += v * s_w3[k * 8 + g];
  }

  float p[8], agc = 0.0f, lbs = 0.0f;
#pragma unroll
  for (int g = 0; g < 8; ++g) {
    p[g] = sigmoid_(logit[g] + s_b[g] + s_g[g]);
    out_probs[b * 8 + g] = p[g];
    agc += (p[g] > 0.5f) ? 1.0f : 0.0f;
    lbs += p[g];
  }

  // stable descending argsort ranks (ties -> lower index first, matches argsort(-p))
  int  rank_[8];
  bool used[8];
#pragma unroll
  for (int g = 0; g < 8; ++g) used[g] = false;
  for (int r = 0; r < 8; ++r) {
    int bi = 0; float bv = -3.4e38f;
#pragma unroll
    for (int g = 0; g < 8; ++g)
      if (!used[g] && p[g] > bv) { bv = p[g]; bi = g; }
    used[bi] = true;
    rank_[bi] = r;
  }
#pragma unroll
  for (int g = 0; g < 8; ++g) {
    float hard = (rank_[g] < 2) ? 1.0f
               : ((rank_[g] < 6) ? ((p[g] > 0.5f) ? 1.0f : 0.0f) : 0.0f);
    out_sel[b * 8 + g] = (hard - p[g]) + p[g];   // straight-through
  }

  sr[0 * 256 + tid] = agc;
#pragma unroll
  for (int g = 0; g < 8; ++g) sr[(1 + g) * 256 + tid] = (p[g] > 0.5f) ? 1.0f : 0.0f;
  sr[9 * 256 + tid] = lbs;
  __syncthreads();
  for (int st = 128; st > 0; st >>= 1) {
    if (tid < st)
#pragma unroll
      for (int v = 0; v < 10; ++v) sr[v * 256 + tid] += sr[v * 256 + tid + st];
    __syncthreads();
  }
  if (tid < 10) pblock[blockIdx.x * 10 + tid] = sr[tid * 256];
}

__global__ void finalize_router_kernel(const float* __restrict__ pblock, int nblocks,
                                       float* __restrict__ out_scal) {
  int v = threadIdx.x;
  if (v >= 10) return;
  float s = 0.0f;
  for (int i = 0; i < nblocks; ++i) s += pblock[i * 10 + v];
  out_scal[v] = s * (1.0f / 32768.0f);   // [0]=active_groups [1..8]=usage [9]=load_balance
}

// ---------------------------------------------------------------------------
// mask = repeat(sel, 96) * dim_weights ; also partials for active_dims
// ---------------------------------------------------------------------------
__global__ __launch_bounds__(256) void mask_kernel(const float* __restrict__ sel,
                                                   const float* __restrict__ dw,
                                                   float* __restrict__ out_mask,
                                                   float* __restrict__ out_gmask,
                                                   float* __restrict__ pm) {
  __shared__ float sc[256];
  long t = (long)blockIdx.x * 256 + threadIdx.x;
  long b = t / 768;
  int  d = (int)(t % 768);
  int  g = d / 96;
  float s = sel[b * 8 + g];
  float w = dw[t];
  float m = s * w;
  out_gmask[t] = s;
  out_mask[t]  = m;
  sc[threadIdx.x] = (m > 0.5f) ? 1.0f : 0.0f;
  __syncthreads();
  for (int st = 128; st > 0; st >>= 1) {
    if (threadIdx.x < st) sc[threadIdx.x] += sc[threadIdx.x + st];
    __syncthreads();
  }
  if (threadIdx.x == 0) pm[blockIdx.x] = sc[0];
}

__global__ __launch_bounds__(256) void finalize_mask_kernel(const float* __restrict__ pm,
                                                            int n, float* __restrict__ out_ad) {
  __shared__ float sc[256];
  float s = 0.0f;
  for (int i = threadIdx.x; i < n; i += 256) s += pm[i];   // fixed order per thread
  sc[threadIdx.x] = s;
  __syncthreads();
  for (int st = 128; st > 0; st >>= 1) {
    if (threadIdx.x < st) sc[threadIdx.x] += sc[threadIdx.x + st];
    __syncthreads();
  }
  if (threadIdx.x == 0) *out_ad = sc[0] * (1.0f / 32768.0f);
}

// ---------------------------------------------------------------------------
// Host driver
// ---------------------------------------------------------------------------
extern "C" void kernel_launch(void* const* d_in, const int* in_sizes, int n_in,
                              void* d_out, int out_size, void* d_ws, size_t ws_size,
                              hipStream_t stream) {
  (void)in_sizes; (void)n_in; (void)out_size; (void)ws_size;
  const long B = 32768, D = 768, G = 8;

  const float* x   = (const float*)d_in[0];
  const float* w1  = (const float*)d_in[1];
  const float* b1  = (const float*)d_in[2];
  const float* lng = (const float*)d_in[3];
  const float* lnb = (const float*)d_in[4];
  const float* w2  = (const float*)d_in[5];
  const float* b2  = (const float*)d_in[6];
  const float* w3  = (const float*)d_in[7];
  const float* b3  = (const float*)d_in[8];
  const float* dw1 = (const float*)d_in[9];
  const float* db1 = (const float*)d_in[10];
  const float* dw2 = (const float*)d_in[11];
  const float* db2 = (const float*)d_in[12];
  const float* gi  = (const float*)d_in[13];

  float* out       = (float*)d_out;
  float* out_mask  = out;
  float* out_gmask = out + B * D;
  float* out_dw    = out + 2 * B * D;
  float* out_probs = out + 3 * B * D;
  float* out_sel   = out + 3 * B * D + B * G;
  float* out_scal  = out + 3 * B * D + 2 * B * G;  // 11 scalars

  char* ws = (char*)d_ws;
  unsigned short* xf   = (unsigned short*)(ws);                        //  48 MB
  unsigned short* h1f  = (unsigned short*)(ws + 50331648L);            //  16 MB
  unsigned short* tf   = (unsigned short*)(ws + 67108864L);            //  16 MB
  float*          ybuf = (float*)(ws + 83886080L);                     //  32 MB
  float*          h2b  = (float*)(ws + 117440512L);                    //  16 MB
  unsigned short* w1f  = (unsigned short*)(ws + 134217728L);
  unsigned short* dw1f = (unsigned short*)(ws + 134217728L + 393216L);
  unsigned short* w2f  = (unsigned short*)(ws + 134217728L + 786432L);
  unsigned short* dw2f = (unsigned short*)(ws + 134217728L + 851968L);
  float*          pblk = (float*)(ws + 134217728L + 1245184L);
  float*          pmsk = pblk + 128 * 10;

  // layout conversions
  cvt_x_kernel<<<(int)((B * D) / 256), 256, 0, stream>>>(x, xf, 24, B * D);
  cvt_w_kernel<<<768 * 256 / 256, 256, 0, stream>>>(w1,  w1f,  256, 768 * 256);
  cvt_w_kernel<<<768 * 256 / 256, 256, 0, stream>>>(dw1, dw1f, 256, 768 * 256);
  cvt_w_kernel<<<256 * 128 / 256, 256, 0, stream>>>(w2,  w2f,  128, 256 * 128);
  cvt_w_kernel<<<256 * 768 / 256, 256, 0, stream>>>(dw2, dw2f, 768, 256 * 768);

  // router path
  gemm_bf16<0, 0><<<dim3(256, 4),  256, 0, stream>>>(xf,  w1f,  b1,  ybuf,   24, 256);
  ln_gelu_kernel<<<(int)(B / 8), 256, 0, stream>>>(ybuf, lng, lnb, h1f);
  gemm_bf16<1, 0><<<dim3(256, 2),  256, 0, stream>>>(h1f, w2f,  b2,  h2b,     8, 128);
  router_final_kernel<<<(int)(B / 256), 256, 0, stream>>>(h2b, w3, b3, gi,
                                                          out_probs, out_sel, pblk);
  finalize_router_kernel<<<1, 32, 0, stream>>>(pblk, (int)(B / 256), out_scal);

  // dim-importance path
  gemm_bf16<1, 1><<<dim3(256, 4),  256, 0, stream>>>(xf, dw1f, db1, tf,      24, 256);
  gemm_bf16<2, 0><<<dim3(256, 12), 256, 0, stream>>>(tf, dw2f, db2, out_dw,   8, 768);

  // combine + active_dims
  mask_kernel<<<(int)((B * D) / 256), 256, 0, stream>>>(out_sel, out_dw,
                                                        out_mask, out_gmask, pmsk);
  finalize_mask_kernel<<<1, 256, 0, stream>>>(pmsk, (int)((B * D) / 256), out_scal + 10);
}
